// CRF_20641612824890
// MI455X (gfx1250) — matrix operand order repaired and checked by reading, not compiled, not run on hardware
//
#include <hip/hip_runtime.h>

#define NTAG 48
#define STARTT 46
#define STOPT 47
#define BB 1024
#define SS 512
#define LOG2E 1.4426950408889634f
#define LN2   0.6931471805599453f
// NEG_INF (-10000) pre-scaled into the log2 domain:
#define NEG_INF2 -14426.950408889634f

typedef __attribute__((ext_vector_type(16))) _Float16 v16h;
typedef __attribute__((ext_vector_type(8)))  float    v8f;

__global__ void zero_out_kernel(float* out) { out[0] = 0.0f; }

// One thread per batch row: gold path score (branchless: mask is a prefix,
// so prev == last and updates can be arithmetic blends).
__global__ void gold_kernel(const float* __restrict__ em,
                            const int* __restrict__ tags,
                            const int* __restrict__ mask,
                            const float* __restrict__ trans,
                            float* __restrict__ gold) {
    int b = blockIdx.x * blockDim.x + threadIdx.x;
    if (b >= BB) return;
    const int*   tg = tags + (size_t)b * SS;
    const int*   mk = mask + (size_t)b * SS;
    const float* eb = em   + (size_t)b * SS * NTAG;
    float score = 0.0f;
    int prev = STARTT;
    for (int t = 0; t < SS; ++t) {
        int   m   = mk[t];
        int   tag = tg[t];
        float contrib = trans[tag * NTAG + prev] + eb[(size_t)t * NTAG + tag];
        score = fmaf((float)m, contrib, score);   // no branch
        prev  = m ? tag : prev;                   // v_cndmask
    }
    score += trans[STOPT * NTAG + prev];
    gold[b] = score;
}

// One wave (32 threads) per 16 batch rows. Exp-domain forward recursion,
// state kept in the log2 domain (beta = alpha * log2(e)):
//   beta'[m][n] = rmax[m] + log2( (exp2A @ expT)[m][n] ) + emit[m][t][n]*LOG2E
// exp2A is (16 x 48) f16 (K padded to 64), expT is (48 x 16) per N-tile.
__launch_bounds__(32)
__global__ void crf_forward_wmma(const float* __restrict__ em,
                                 const int* __restrict__ mask,
                                 const float* __restrict__ trans,
                                 const float* __restrict__ gold,
                                 float* __restrict__ out) {
    // Double-buffered transpose staging: removes the WAR barrier per step.
    __shared__ float sexp[2][16 * 64];   // [buf][row m][K padded to 64]

    const int lane = threadIdx.x;     // 0..31
    const int g    = lane >> 4;       // half-wave group
    const int ln   = lane & 15;
    const int b0   = blockIdx.x * 16;

    // Zero both buffers once (pad columns 48..63 stay zero forever).
    for (int i = lane; i < 2 * 16 * 64; i += 32) (&sexp[0][0])[i] = 0.0f;
    __syncthreads();

    // ---- B fragments: exp(transitions) (linear domain), K padded to 64 ----
    // B 32x16 f16 layout: lane -> N = ln, half h -> K = 8*g + h (h<8), 16 + 8*g + (h-8) (h>=8) (per 32-K tile)
    v16h bfrag[2][3];
#pragma unroll
    for (int kt = 0; kt < 2; ++kt) {
#pragma unroll
        for (int c = 0; c < 3; ++c) {
#pragma unroll
            for (int h = 0; h < 16; ++h) {
                int K = 32 * kt + 8 * g + (h < 8 ? h : h + 8);
                int n = ln + 16 * c;
                float v = (K < NTAG)
                    ? __builtin_amdgcn_exp2f(trans[K * NTAG + n] * LOG2E) : 0.0f;
                bfrag[kt][c][h] = (_Float16)v;
            }
        }
    }

    // ---- sequence lengths: lane computes len of row (lane&15) ----
    int mylen = 0;
    {
        const int4* mrow = (const int4*)(mask + (size_t)(b0 + ln) * SS);
        for (int t = 0; t < SS / 4; ++t) {
            int4 v = mrow[t];
            mylen += v.x + v.y + v.z + v.w;
        }
    }
    // C-layout: element r of a C fragment holds row m = r + 8*g
    int len8[8];
#pragma unroll
    for (int r = 0; r < 8; ++r) len8[r] = __shfl(mylen, r + 8 * g, 32);

    // ---- init beta (C layout in registers): -inf except START col ----
    float beta[8][3];
#pragma unroll
    for (int r = 0; r < 8; ++r)
#pragma unroll
        for (int c = 0; c < 3; ++c)
            beta[r][c] = ((ln + 16 * c) == STARTT) ? 0.0f : NEG_INF2;

    const float* embase = em + (size_t)b0 * SS * NTAG;

    // Emission prefetch helper (24 b32 loads, each element touched once).
    auto load_ev = [&](int t, float ev[8][3]) {
#pragma unroll
        for (int r = 0; r < 8; ++r) {
            const float* ep = embase + ((size_t)(r + 8 * g) * SS + t) * NTAG + ln;
#pragma unroll
            for (int c = 0; c < 3; ++c) ev[r][c] = ep[16 * c];
        }
    };

    // One recursion step using already-loaded emissions for time t.
    auto step = [&](int t, float* sb, const float ev[8][3]) {
        // Row max (reduce over 3 frags in-lane + 16 lanes of the group).
        float rmax[8];
#pragma unroll
        for (int r = 0; r < 8; ++r) {
            float m = fmaxf(fmaxf(beta[r][0], beta[r][1]), beta[r][2]);
#pragma unroll
            for (int s = 1; s < 16; s <<= 1) m = fmaxf(m, __shfl_xor(m, s, 32));
            rmax[r] = m;
        }

        // Store exp2(beta - rmax) to LDS in row-major [m][K].
#pragma unroll
        for (int r = 0; r < 8; ++r) {
            int m = r + 8 * g;
#pragma unroll
            for (int c = 0; c < 3; ++c)
                sb[m * 64 + ln + 16 * c] =
                    __builtin_amdgcn_exp2f(beta[r][c] - rmax[r]);
        }
        __syncthreads();   // single-wave block: lowers to wave barrier + dscnt

        // A fragments (16x32 f16 per K-tile): lane -> row ln,
        // halves 0..7 -> K = 32*kt + 8*g + h, halves 8..15 -> +16.
        v16h afrag[2];
#pragma unroll
        for (int kt = 0; kt < 2; ++kt) {
            const float* base = &sb[ln * 64 + 32 * kt + 8 * g];
            float4 v0 = *(const float4*)(base + 0);
            float4 v1 = *(const float4*)(base + 4);
            float4 v2 = *(const float4*)(base + 16);
            float4 v3 = *(const float4*)(base + 20);
            afrag[kt][0]  = (_Float16)v0.x; afrag[kt][1]  = (_Float16)v0.y;
            afrag[kt][2]  = (_Float16)v0.z; afrag[kt][3]  = (_Float16)v0.w;
            afrag[kt][4]  = (_Float16)v1.x; afrag[kt][5]  = (_Float16)v1.y;
            afrag[kt][6]  = (_Float16)v1.z; afrag[kt][7]  = (_Float16)v1.w;
            afrag[kt][8]  = (_Float16)v2.x; afrag[kt][9]  = (_Float16)v2.y;
            afrag[kt][10] = (_Float16)v2.z; afrag[kt][11] = (_Float16)v2.w;
            afrag[kt][12] = (_Float16)v3.x; afrag[kt][13] = (_Float16)v3.y;
            afrag[kt][14] = (_Float16)v3.z; afrag[kt][15] = (_Float16)v3.w;
        }
        // No WAR barrier: next step stores into the other buffer.

        // (16x64) @ (64x48) via 6 WMMAs.
        v8f csum[3];
#pragma unroll
        for (int c = 0; c < 3; ++c) {
            v8f acc = {};
            acc = __builtin_amdgcn_wmma_f32_16x16x32_f16(
                false, afrag[0], false, bfrag[0][c], (short)0, acc, false, false);
            acc = __builtin_amdgcn_wmma_f32_16x16x32_f16(
                false, afrag[1], false, bfrag[1][c], (short)0, acc, false, false);
            csum[c] = acc;
        }

        // Branchless masked update; s+1e-10 replaces clamp (no canonicalize,
        // same value wherever it matters) and keeps beta finite for the blend.
#pragma unroll
        for (int r = 0; r < 8; ++r) {
            float lf   = (t < len8[r]) ? 1.0f : 0.0f;   // v_cndmask
            float omlf = 1.0f - lf;
#pragma unroll
            for (int c = 0; c < 3; ++c) {
                float s    = csum[c][r] + 1e-10f;
                float bnew = fmaf(ev[r][c], LOG2E, rmax[r])
                           + __builtin_amdgcn_logf(s);
                beta[r][c] = fmaf(lf, bnew, omlf * beta[r][c]);
            }
        }
    };

    // 2x-unrolled, software-pipelined scan: prefetch t+1 while computing t.
    float evA[8][3], evB[8][3];
    load_ev(0, evA);
    for (int t = 0; t < SS; t += 2) {
        load_ev(t + 1, evB);                         // t+1 <= SS-1 always
        step(t, &sexp[0][0], evA);
        int t2 = (t + 2 < SS) ? (t + 2) : (SS - 1);  // last prefetch harmless
        load_ev(t2, evA);
        step(t + 1, &sexp[1][0], evB);
    }

    // ---- terminal: + trans[STOP][n]*LOG2E, then logsumexp (base 2) ----
#pragma unroll
    for (int c = 0; c < 3; ++c) {
        float tstop = trans[STOPT * NTAG + ln + 16 * c] * LOG2E;
#pragma unroll
        for (int r = 0; r < 8; ++r) beta[r][c] += tstop;
    }

    float logz[8];
#pragma unroll
    for (int r = 0; r < 8; ++r) {
        float m = fmaxf(fmaxf(beta[r][0], beta[r][1]), beta[r][2]);
#pragma unroll
        for (int s = 1; s < 16; s <<= 1) m = fmaxf(m, __shfl_xor(m, s, 32));
        float sum = __builtin_amdgcn_exp2f(beta[r][0] - m) +
                    __builtin_amdgcn_exp2f(beta[r][1] - m) +
                    __builtin_amdgcn_exp2f(beta[r][2] - m);
#pragma unroll
        for (int s = 1; s < 16; s <<= 1) sum += __shfl_xor(sum, s, 32);
        logz[r] = (m + __builtin_amdgcn_logf(sum + 1e-10f)) * LN2;
    }

    // Lane ln==r writes row r + 8*g.
#pragma unroll
    for (int r = 0; r < 8; ++r) {
        if (ln == r) {
            int b = b0 + r + 8 * g;
            float contrib = (logz[r] - gold[b]) * (1.0f / (float)BB);
            atomicAdd(out, contrib);
        }
    }
}

extern "C" void kernel_launch(void* const* d_in, const int* in_sizes, int n_in,
                              void* d_out, int out_size, void* d_ws, size_t ws_size,
                              hipStream_t stream) {
    const float* em    = (const float*)d_in[0];   // (1024,512,48) f32
    const int*   tags  = (const int*)  d_in[1];   // (1024,512) i32
    const int*   mask  = (const int*)  d_in[2];   // (1024,512) i32
    const float* trans = (const float*)d_in[3];   // (48,48) f32
    float* out  = (float*)d_out;                  // scalar
    float* gold = (float*)d_ws;                   // 1024 floats of scratch

    zero_out_kernel<<<1, 1, 0, stream>>>(out);
    gold_kernel<<<BB / 256, 256, 0, stream>>>(em, tags, mask, trans, gold);
    crf_forward_wmma<<<BB / 16, 32, 0, stream>>>(em, mask, trans, gold, out);
}
